// GSAMechanism_31817117729217
// MI455X (gfx1250) — compile-verified, matrix operand order checked by reading, not run on hardware
//
#include <hip/hip_runtime.h>
#include <hip/hip_bf16.h>
#include <cstdint>

typedef __attribute__((ext_vector_type(16))) __bf16 v16bf;
typedef __attribute__((ext_vector_type(8)))  __bf16 v8bf;
typedef __attribute__((ext_vector_type(8)))  float  v8f;
typedef __attribute__((ext_vector_type(2)))  float  v2f;
typedef int v4i_vs __attribute__((vector_size(16)));   // matches builtin param type

#define B_  2
#define T_  2048
#define D_  1024
#define H_  16
#define S_  16
#define HD_ 64

// ---------------------------------------------------------------------------
// CDNA5 async global->LDS copy (ASYNCcnt path), guarded so either toolchain
// compiles. Fallback = 128-bit load/store through VGPRs.
// ---------------------------------------------------------------------------
#if defined(__AMDGCN__) && __has_builtin(__builtin_amdgcn_global_load_async_to_lds_b128)
#define USE_ASYNC_LDS 1
#else
#define USE_ASYNC_LDS 0
#endif

#if USE_ASYNC_LDS
__device__ inline void async_b128(const void* g, void* l) {
    __builtin_amdgcn_global_load_async_to_lds_b128((v4i_vs*)g, (v4i_vs*)l, 0, 0);
}
// wait until at most N of this wave's async ops remain outstanding
#define ASYNC_WAIT(N)                                                \
    do {                                                             \
        _Pragma("clang diagnostic push")                             \
        __builtin_amdgcn_s_wait_asynccnt(N);                         \
        _Pragma("clang diagnostic pop")                              \
    } while (0)
#endif

// Build a 16-bf16 WMMA operand from two contiguous 8-element runs.
__device__ inline v16bf frag16(const __bf16* lo, const __bf16* hi) {
    v8bf a = *(const v8bf*)lo;
    v8bf b = *(const v8bf*)hi;
    return __builtin_shufflevector(a, b, 0, 1, 2, 3, 4, 5, 6, 7,
                                   8, 9, 10, 11, 12, 13, 14, 15);
}

// ---------------------------------------------------------------------------
// fp32 -> bf16 elementwise convert
// ---------------------------------------------------------------------------
__global__ __launch_bounds__(256)
void cvt_bf16_kernel(const float* __restrict__ in, __bf16* __restrict__ out, int n) {
    int g = blockIdx.x * 256 + threadIdx.x;
    if (g < n) out[g] = (__bf16)in[g];
}

// ---------------------------------------------------------------------------
// bf16 GEMM: C[M,N] = A[M,K] * W[N,K]^T   (fp32 accumulate via WMMA)
// block = 256 threads (8 waves), tile 128(M) x 64(N), K step 32.
// Double-buffered LDS: async copy engine streams slab k+1 while the matrix
// pipe consumes slab k (s_wait_asynccnt 3 keeps one 3-op group in flight).
// ---------------------------------------------------------------------------
__global__ __launch_bounds__(256)
void gemm_bf16_kernel(const __bf16* __restrict__ A,
                      const __bf16* __restrict__ W,
                      float* __restrict__ C,
                      int M, int N, int K) {
    __shared__ alignas(16) __bf16 As[2][128 * 32];
    __shared__ alignas(16) __bf16 Ws[2][64 * 32];

    const int tid  = threadIdx.x;
    const int wave = tid >> 5;
    const int lane = tid & 31;
    const int half = lane >> 4;
    const int l16  = lane & 15;
    const int m0   = blockIdx.y * 128;
    const int n0   = blockIdx.x * 64;
    const int NK   = K / 32;

    // per-thread staging coordinates (3 x b128 per thread per slab)
    const size_t ga0 = (size_t)(m0 + (tid >> 2)) * K + (tid & 3) * 8;
    const size_t ga1 = (size_t)(m0 + ((tid + 256) >> 2)) * K + (tid & 3) * 8;
    const size_t gw0 = (size_t)(n0 + (tid >> 2)) * K + (tid & 3) * 8;

    v8f acc[4] = {};

#if USE_ASYNC_LDS
    // prologue: stage slab 0 into buffer 0
    async_b128(A + ga0, (char*)As[0] + tid * 16);
    async_b128(A + ga1, (char*)As[0] + (tid + 256) * 16);
    async_b128(W + gw0, (char*)Ws[0] + tid * 16);
#endif

    for (int kc = 0; kc < NK; ++kc) {
        const int cur = kc & 1;
        const int k0  = kc * 32;
#if USE_ASYNC_LDS
        if (kc + 1 < NK) {
            const int nxt = cur ^ 1;
            const int kn  = k0 + 32;
            async_b128(A + ga0 + kn, (char*)As[nxt] + tid * 16);
            async_b128(A + ga1 + kn, (char*)As[nxt] + (tid + 256) * 16);
            async_b128(W + gw0 + kn, (char*)Ws[nxt] + tid * 16);
            ASYNC_WAIT(3);          // slab kc landed; slab kc+1 still streaming
        } else {
            ASYNC_WAIT(0);
        }
        __syncthreads();
#else
        __syncthreads();
        #pragma unroll
        for (int i = 0; i < 2; ++i) {
            int idx = tid + i * 256;
            ((uint4*)As[cur])[idx] =
                *(const uint4*)(A + (size_t)(m0 + (idx >> 2)) * K + k0 + (idx & 3) * 8);
        }
        ((uint4*)Ws[cur])[tid] =
            *(const uint4*)(W + (size_t)(n0 + (tid >> 2)) * K + k0 + (tid & 3) * 8);
        if (k0 + 32 < K) {
            __builtin_prefetch(A + (size_t)(m0 + (tid >> 1)) * K + k0 + 32, 0, 1);
            __builtin_prefetch(W + (size_t)(n0 + (tid & 63)) * K + k0 + 32, 0, 1);
        }
        __syncthreads();
#endif

        // A fragment: row = wave*16 + l16. Half 0 holds K 0-7|16-23, half 1 K 8-15|24-31.
        const __bf16* arow = As[cur] + (wave * 16 + l16) * 32;
        v16bf a = frag16(arow + 8 * half, arow + 16 + 8 * half);
        #pragma unroll
        for (int nt = 0; nt < 4; ++nt) {
            const __bf16* brow = Ws[cur] + (nt * 16 + l16) * 32;
            v16bf bfr = frag16(brow + 8 * half, brow + 16 + 8 * half);
            acc[nt] = __builtin_amdgcn_wmma_f32_16x16x32_bf16(
                false, a, false, bfr, (short)0, acc[nt], false, false);
        }
#if USE_ASYNC_LDS
        __syncthreads();   // all waves done reading buf[cur] before it is re-filled
#endif
    }

    #pragma unroll
    for (int nt = 0; nt < 4; ++nt) {
        #pragma unroll
        for (int r = 0; r < 8; ++r) {
            int row = m0 + wave * 16 + r + 8 * half;
            int col = n0 + nt * 16 + l16;
            C[(size_t)row * N + col] = acc[nt][r];
        }
    }
}

// ---------------------------------------------------------------------------
// Gaussian splat weights (amp folded into qw)
// qkv (B,T,3,H,HD) fp32 ; outputs (B,H,T,S) fp32
// ---------------------------------------------------------------------------
__global__ __launch_bounds__(256)
void splat_weights_kernel(const float* __restrict__ qkv,
                          const float* __restrict__ centers,
                          const float* __restrict__ deltas,
                          const float* __restrict__ log_scales,
                          const float* __restrict__ log_amps,
                          const float* __restrict__ movement_scale,
                          float* __restrict__ qw, float* __restrict__ kw) {
    int g = blockIdx.x * 256 + threadIdx.x;
    if (g >= B_ * T_ * H_ * S_) return;
    int s = g & 15;
    int r = g >> 4;
    int h = r & 15;  r >>= 4;
    int t = r & (T_ - 1);
    int b = r >> 11;

    float ms   = movement_scale[0];
    float coef = (1.0f / (1.0f + __expf(-ms))) * 0.2f;
    int hs = h * S_ + s;
    float scale = __expf(log_scales[hs]);
    scale = fminf(fmaxf(scale, 0.01f), 2.0f);
    float inv_var = -0.5f / (scale * scale + 1e-8f);
    float amp = __expf(log_amps[hs]);
    amp = fminf(fmaxf(amp, 1e-6f), 10.0f);

    size_t qbase = ((size_t)(b * T_ + t)) * (3 * D_) + h * HD_;
    const float4* qp = (const float4*)(qkv + qbase);
    const float4* kp = (const float4*)(qkv + qbase + D_);
    const float4* cp = (const float4*)(centers + (size_t)hs * HD_);
    const float4* dp = (const float4*)(deltas + (size_t)hs * HD_);

    float dq = 0.f, dk = 0.f;
    #pragma unroll 4
    for (int d = 0; d < HD_ / 4; ++d) {
        float4 c4 = cp[d], dl = dp[d], q4 = qp[d], k4 = kp[d];
        float cx = c4.x + dl.x * coef, cy = c4.y + dl.y * coef;
        float cz = c4.z + dl.z * coef, cw = c4.w + dl.w * coef;
        float v;
        v = q4.x - cx; dq += v * v;  v = k4.x - cx; dk += v * v;
        v = q4.y - cy; dq += v * v;  v = k4.y - cy; dk += v * v;
        v = q4.z - cz; dq += v * v;  v = k4.z - cz; dk += v * v;
        v = q4.w - cw; dq += v * v;  v = k4.w - cw; dk += v * v;
    }
    size_t o = ((size_t)(b * H_ + h) * T_ + t) * S_ + s;
    qw[o] = __expf(dq * inv_var) * amp;
    kw[o] = __expf(dk * inv_var);
}

// ---------------------------------------------------------------------------
// Repack V plane of qkv (fp32, (B,T,3,H,HD)) -> bf16 TRANSPOSED (B,H,HD,T)
// so attention P*V B-fragments are contiguous 16B runs along K (=t).
// ---------------------------------------------------------------------------
__global__ __launch_bounds__(256)
void v_repack_kernel(const float* __restrict__ qkv, __bf16* __restrict__ vT) {
    int g = blockIdx.x * 256 + threadIdx.x;
    if (g >= B_ * H_ * HD_ * T_) return;
    int t = g & (T_ - 1);
    int r = g >> 11;
    int d = r & (HD_ - 1); r >>= 6;
    int h = r & 15;
    int b = r >> 4;
    vT[g] = (__bf16)qkv[((size_t)(b * T_ + t)) * (3 * D_) + 2 * D_ + h * HD_ + d];
}

// ---------------------------------------------------------------------------
// Attention: one block per (b, h, 16-query tile). 128 threads = 4 waves.
// Phase 1: logits 16x2048 via exact v_wmma_f32_16x16x4_f32 (K=S=16) -> LDS
// Phase 2: row softmax (scaled by 1/temp) in LDS, write bf16 probs
// Phase 3: P(16x2048 bf16, LDS) x V^T(global, (HD,T)) via bf16 WMMA
// Dynamic LDS: 16*2048*4 + 16*2048*2 = 192 KB
// ---------------------------------------------------------------------------
__global__ __launch_bounds__(128)
void attention_kernel(const float* __restrict__ qw,     // (B,H,T,S) amp folded
                      const float* __restrict__ kw,     // (B,H,T,S)
                      const __bf16* __restrict__ vT,    // (B,H,HD,T)
                      const float* __restrict__ temperature,
                      float* __restrict__ attn_out) {   // (B,T,H,HD)
    extern __shared__ char smem[];
    float*  Pf = (float*)smem;                       // 16 x 2048 logits
    __bf16* Pb = (__bf16*)(smem + 16 * T_ * 4);      // 16 x 2048 probs (bf16)
    __shared__ float red[16 * 8];
    __shared__ float rowstat[16];

    const int tid  = threadIdx.x;
    const int wave = tid >> 5;
    const int lane = tid & 31;
    const int half = lane >> 4;
    const int l16  = lane & 15;

    int blk = blockIdx.x;               // B*H*(T/16) = 4096
    int it  = blk & 127;
    int h   = (blk >> 7) & 15;
    int b   = blk >> 11;
    const size_t bh = (size_t)b * H_ + h;
    const float*  qwp = qw + bh * T_ * S_;
    const float*  kwp = kw + bh * T_ * S_;
    const __bf16* vp  = vT + bh * HD_ * T_;
    const int i0 = it * 16;

    // ---- Phase 1: logits via f32 WMMA (16x16x4, 4 k-steps cover S=16) ----
    v2f afrag[4];
    #pragma unroll
    for (int ks = 0; ks < 4; ++ks)
        afrag[ks] = *(const v2f*)(qwp + (size_t)(i0 + l16) * S_ + ks * 4 + 2 * half);

    for (int j = wave; j < T_ / 16; j += 4) {
        v8f acc = {};
        #pragma unroll
        for (int ks = 0; ks < 4; ++ks) {
            v2f bfrag = *(const v2f*)(kwp + (size_t)(j * 16 + l16) * S_ + ks * 4 + 2 * half);
            acc = __builtin_amdgcn_wmma_f32_16x16x4_f32(
                false, afrag[ks], false, bfrag, (short)0, acc, false, false);
        }
        #pragma unroll
        for (int r = 0; r < 8; ++r)
            Pf[(r + 8 * half) * T_ + j * 16 + l16] = acc[r];
    }
    __syncthreads();

    // ---- Phase 2: softmax over each of the 16 rows (2048 cols) ----
    float tt = temperature[0];
    tt = fminf(fmaxf(tt, 0.1f), 10.0f);
    const float invt = 1.0f / tt;

    const int row   = tid & 15;
    const int chunk = tid >> 4;                   // 8 chunks of 256 cols
    const int c0    = chunk * 256;

    float m = -3.4e38f;
    for (int c = c0; c < c0 + 256; ++c) m = fmaxf(m, Pf[row * T_ + c]);
    red[row * 8 + chunk] = m;
    __syncthreads();
    if (tid < 16) {
        float mm = red[tid * 8];
        #pragma unroll
        for (int i = 1; i < 8; ++i) mm = fmaxf(mm, red[tid * 8 + i]);
        rowstat[tid] = mm;
    }
    __syncthreads();
    float rm = rowstat[row];
    float sum = 0.f;
    for (int c = c0; c < c0 + 256; ++c) {
        float p = __expf((Pf[row * T_ + c] - rm) * invt);
        Pf[row * T_ + c] = p;
        sum += p;
    }
    __syncthreads();
    red[row * 8 + chunk] = sum;
    __syncthreads();
    if (tid < 16) {
        float ss = 0.f;
        #pragma unroll
        for (int i = 0; i < 8; ++i) ss += red[tid * 8 + i];
        rowstat[tid] = ss;
    }
    __syncthreads();
    float inv = 1.0f / rowstat[row];
    for (int c = c0; c < c0 + 256; ++c)
        Pb[row * T_ + c] = (__bf16)(Pf[row * T_ + c] * inv);
    __syncthreads();

    // ---- Phase 3: P x V, wave w owns HD columns [w*16, w*16+16) ----
    const int d0 = wave * 16;
    const __bf16* prow = Pb + l16 * T_;                 // A row (query l16)
    const __bf16* vrow = vp + (size_t)(d0 + l16) * T_;  // B "column" = V^T row
    v8f acc = {};
    for (int kc = 0; kc < T_ / 32; ++kc) {
        v16bf a  = frag16(prow + kc * 32 + 8 * half, prow + kc * 32 + 16 + 8 * half);
        v16bf bb = frag16(vrow + kc * 32 + 8 * half, vrow + kc * 32 + 16 + 8 * half);
        acc = __builtin_amdgcn_wmma_f32_16x16x32_bf16(
            false, a, false, bb, (short)0, acc, false, false);
    }
    #pragma unroll
    for (int r = 0; r < 8; ++r) {
        int qrow = i0 + r + 8 * half;
        attn_out[((size_t)(b * T_ + qrow)) * D_ + h * HD_ + d0 + l16] = acc[r];
    }
}

// ---------------------------------------------------------------------------
// Host-side launcher
// ---------------------------------------------------------------------------
extern "C" void kernel_launch(void* const* d_in, const int* in_sizes, int n_in,
                              void* d_out, int out_size, void* d_ws, size_t ws_size,
                              hipStream_t stream) {
    (void)in_sizes; (void)n_in; (void)out_size; (void)ws_size;

    const float* x        = (const float*)d_in[0];
    const float* centers  = (const float*)d_in[1];
    const float* deltas   = (const float*)d_in[2];
    const float* lscales  = (const float*)d_in[3];
    const float* lamps    = (const float*)d_in[4];
    const float* mscale   = (const float*)d_in[5];
    const float* temp     = (const float*)d_in[6];
    const float* qkv_w    = (const float*)d_in[7];
    const float* out_w    = (const float*)d_in[8];
    float* out = (float*)d_out;

    const int BT = B_ * T_;                 // 4096
    char* ws = (char*)d_ws;
    size_t off = 0;
    __bf16* xbf     = (__bf16*)(ws + off); off += (size_t)BT * D_ * 2;           // 8 MB
    __bf16* qkvwbf  = (__bf16*)(ws + off); off += (size_t)3 * D_ * D_ * 2;       // 6 MB
    __bf16* outwbf  = (__bf16*)(ws + off); off += (size_t)D_ * D_ * 2;           // 2 MB
    float*  qkv     = (float*) (ws + off); off += (size_t)BT * 3 * D_ * 4;       // 48 MB
    float*  qw      = (float*) (ws + off); off += (size_t)B_ * H_ * T_ * S_ * 4; // 4 MB
    float*  kw      = (float*) (ws + off); off += (size_t)B_ * H_ * T_ * S_ * 4; // 4 MB
    __bf16* vT      = (__bf16*)(ws + off); off += (size_t)B_ * H_ * HD_ * T_ * 2;// 8 MB
    float*  attn    = (float*) (ws + off); off += (size_t)BT * D_ * 4;           // 16 MB
    __bf16* attnbf  = (__bf16*)(ws + off); off += (size_t)BT * D_ * 2;           // 8 MB

    // 1) bf16 conversions of x / weights
    {
        int n = BT * D_;
        cvt_bf16_kernel<<<(n + 255) / 256, 256, 0, stream>>>(x, xbf, n);
        n = 3 * D_ * D_;
        cvt_bf16_kernel<<<(n + 255) / 256, 256, 0, stream>>>(qkv_w, qkvwbf, n);
        n = D_ * D_;
        cvt_bf16_kernel<<<(n + 255) / 256, 256, 0, stream>>>(out_w, outwbf, n);
    }

    // 2) QKV projection: (4096 x 1024) x (3072 x 1024)^T
    {
        dim3 grid((3 * D_) / 64, BT / 128);
        gemm_bf16_kernel<<<grid, 256, 0, stream>>>(xbf, qkvwbf, qkv, BT, 3 * D_, D_);
    }

    // 3) Splat weights (amp folded into qw)
    {
        int n = B_ * T_ * H_ * S_;
        splat_weights_kernel<<<(n + 255) / 256, 256, 0, stream>>>(
            qkv, centers, deltas, lscales, lamps, mscale, qw, kw);
    }

    // 4) V repack to bf16 transposed (B,H,HD,T)
    {
        int n = B_ * H_ * HD_ * T_;
        v_repack_kernel<<<(n + 255) / 256, 256, 0, stream>>>(qkv, vT);
    }

    // 5) Attention (logits f32-WMMA + LDS softmax + P*V bf16-WMMA)
    {
        int blocks = B_ * H_ * (T_ / 16);
        size_t lds = (size_t)16 * T_ * 4 + (size_t)16 * T_ * 2;   // 192 KB
        attention_kernel<<<blocks, 128, lds, stream>>>(qw, kw, vT, temp, attn);
    }

    // 6) Output projection: (4096 x 1024) x (1024 x 1024)^T
    {
        int n = BT * D_;
        cvt_bf16_kernel<<<(n + 255) / 256, 256, 0, stream>>>(attn, attnbf, n);
        dim3 grid(D_ / 64, BT / 128);
        gemm_bf16_kernel<<<grid, 256, 0, stream>>>(attnbf, outwbf, out, BT, D_, D_);
    }
}
